// GIN_91122026152449
// MI455X (gfx1250) — compile-verified, matrix operand order checked by reading, not run on hardware
//
#include <hip/hip_runtime.h>

#define HD 128  // D == H == 128

typedef __attribute__((ext_vector_type(16))) __bf16 v16bf;
typedef __attribute__((ext_vector_type(8)))  float  v8f;

union Frag16 { uint4 q[2]; v16bf v; };

__device__ __forceinline__ unsigned short f2bf(float f) {
  unsigned int u = __float_as_uint(f);
  u += 0x7FFFu + ((u >> 16) & 1u);   // round to nearest even
  return (unsigned short)(u >> 16);
}

// ---------------- utility ----------------
__global__ void k_zero4(float4* __restrict__ p, long long n4) {
  long long i = (long long)blockIdx.x * blockDim.x + threadIdx.x;
  if (i < n4) p[i] = make_float4(0.f, 0.f, 0.f, 0.f);
}

// transpose + cast weights: w[l][k][j] (Ld,K,J) -> wt[l][j][k] bf16
__global__ void k_wt_bf16(const float* __restrict__ w, unsigned short* __restrict__ wt,
                          int Ld, int K, int J) {
  int idx = blockIdx.x * blockDim.x + threadIdx.x;
  int total = Ld * K * J;
  if (idx >= total) return;
  int l = idx / (K * J);
  int rem = idx % (K * J);
  int k = rem / J;
  int j = rem % J;
  wt[((long long)l * J + j) * K + k] = f2bf(w[idx]);
}

// ---------------- edge scatter-add: one wave per edge ----------------
__global__ void k_scatter(const float* __restrict__ h, const int* __restrict__ src,
                          const int* __restrict__ dst, float* __restrict__ agg, int E) {
  int gid = blockIdx.x * blockDim.x + threadIdx.x;
  int e = gid >> 5;
  int lane = gid & 31;
  if (e >= E) return;
  int s = src[e], d = dst[e];
  float4 v = ((const float4*)(h + (long long)s * HD))[lane];
  float* a = agg + (long long)d * HD + lane * 4;
  atomicAdd(a + 0, v.x);
  atomicAdd(a + 1, v.y);
  atomicAdd(a + 2, v.z);
  atomicAdd(a + 3, v.w);
}

// pooled = agg + (1+eps)*h  -> bf16   (4 elements / thread)
__global__ void k_pooled_bf16(const float4* __restrict__ agg, const float4* __restrict__ h,
                              const float* __restrict__ eps, int l,
                              ushort4* __restrict__ out, long long n4) {
  long long i = (long long)blockIdx.x * blockDim.x + threadIdx.x;
  if (i >= n4) return;
  float s = 1.0f + eps[l];
  float4 a = agg[i], b = h[i];
  ushort4 o;
  o.x = f2bf(fmaf(s, b.x, a.x));
  o.y = f2bf(fmaf(s, b.y, a.y));
  o.z = f2bf(fmaf(s, b.z, a.z));
  o.w = f2bf(fmaf(s, b.w, a.w));
  out[i] = o;
}

// ---------------- WMMA GEMM: C(N x 128) = A(N x 128) @ Bt^T + bias ----------------
// A bf16 row-major; Bt = B transposed (128_out x 128_in) bf16.
// One wave per 64-row x 16-col strip: B fragments loaded once, reused over 4 M-tiles.
__global__ void k_gemm128(const unsigned short* __restrict__ A,
                          const unsigned short* __restrict__ Bt,
                          const float* __restrict__ bias,
                          float* __restrict__ C, int nrows) {
  int lane = threadIdx.x & 31;
  int r  = lane & 15;
  int hi = lane >> 4;
  int row0 = blockIdx.x * 64;
  int col0 = blockIdx.y * 16;

  // B fragments for all 4 K-steps (lanes 0-15: K=[0,8)+[16,24); 16-31: K=[8,16)+[24,32))
  const unsigned short* brow = Bt + (long long)(col0 + r) * HD;
  Frag16 b[4];
#pragma unroll
  for (int ks = 0; ks < 4; ++ks) {
    int k0 = ks * 32 + hi * 8;
    b[ks].q[0] = *(const uint4*)(brow + k0);
    b[ks].q[1] = *(const uint4*)(brow + k0 + 16);
  }

  int n = lane & 15;
  float bv = bias[col0 + n];

#pragma unroll
  for (int mt = 0; mt < 4; ++mt) {
    int rbase = row0 + mt * 16;
    if (rbase >= nrows) break;                      // uniform branch
    int arow_i = min(rbase + r, nrows - 1);
    const unsigned short* arow = A + (long long)arow_i * HD;
    v8f c = {};
#pragma unroll
    for (int ks = 0; ks < 4; ++ks) {
      int k0 = ks * 32 + hi * 8;
      Frag16 a;
      a.q[0] = *(const uint4*)(arow + k0);
      a.q[1] = *(const uint4*)(arow + k0 + 16);
      c = __builtin_amdgcn_wmma_f32_16x16x32_bf16(false, a.v, false, b[ks].v,
                                                  (short)0, c, false, false);
    }
    // C layout: VGPR i -> M = i + 8*(lane>=16), N = lane&15
    float* crow = C + (long long)rbase * HD + col0 + n;
    if (rbase + 16 <= nrows) {                      // uniform full-tile fast path
#pragma unroll
      for (int i = 0; i < 8; ++i) {
        int m = i + hi * 8;
        crow[(long long)m * HD] = c[i] + bv;
      }
    } else {
#pragma unroll
      for (int i = 0; i < 8; ++i) {
        int m = i + hi * 8;
        if (rbase + m < nrows) crow[(long long)m * HD] = c[i] + bv;
      }
    }
  }
}

// ---------------- BatchNorm: column stats over node axis ----------------
__global__ void k_bnstats(const float* __restrict__ z, float* __restrict__ sum,
                          float* __restrict__ sumsq, int nrows, int rows_per_block) {
  int j  = threadIdx.x;                         // 128 threads, one per column
  int r0 = blockIdx.x * rows_per_block;
  int r1 = min(r0 + rows_per_block, nrows);
  float s = 0.f, ss = 0.f;
  for (int i = r0; i < r1; ++i) {
    float v = z[(long long)i * HD + j];
    s += v;
    ss = fmaf(v, v, ss);
  }
  atomicAdd(&sum[j], s);
  atomicAdd(&sumsq[j], ss);
}

__global__ void k_bnfin(const float* __restrict__ sum, const float* __restrict__ sumsq,
                        const float* __restrict__ gamma, const float* __restrict__ beta,
                        float* __restrict__ scale, float* __restrict__ shift, float invN) {
  int j = threadIdx.x;
  float m  = sum[j] * invN;
  float v  = fmaxf(sumsq[j] * invN - m * m, 0.0f);  // biased variance
  float is = rsqrtf(v + 1e-5f);
  float sc = gamma[j] * is;
  scale[j] = sc;
  shift[j] = beta[j] - m * sc;
}

// BN+ReLU -> bf16  (4 elements / thread; 128 % 4 == 0 so the quad shares no wrap)
__global__ void k_bnrelu_bf16(const float4* __restrict__ z, const float* __restrict__ scale,
                              const float* __restrict__ shift, ushort4* __restrict__ out,
                              long long n4) {
  long long i = (long long)blockIdx.x * blockDim.x + threadIdx.x;
  if (i >= n4) return;
  int j = (int)((i * 4) & (HD - 1));
  float4 sc = *(const float4*)(scale + j);
  float4 sh = *(const float4*)(shift + j);
  float4 v = z[i];
  ushort4 o;
  o.x = f2bf(fmaxf(fmaf(v.x, sc.x, sh.x), 0.0f));
  o.y = f2bf(fmaxf(fmaf(v.y, sc.y, sh.y), 0.0f));
  o.z = f2bf(fmaxf(fmaf(v.z, sc.z, sh.z), 0.0f));
  o.w = f2bf(fmaxf(fmaf(v.w, sc.w, sh.w), 0.0f));
  out[i] = o;
}

// BN+ReLU -> f32  (4 elements / thread)
__global__ void k_bnrelu_f32(const float4* __restrict__ z, const float* __restrict__ scale,
                             const float* __restrict__ shift, float4* __restrict__ out,
                             long long n4) {
  long long i = (long long)blockIdx.x * blockDim.x + threadIdx.x;
  if (i >= n4) return;
  int j = (int)((i * 4) & (HD - 1));
  float4 sc = *(const float4*)(scale + j);
  float4 sh = *(const float4*)(shift + j);
  float4 v = z[i];
  float4 o;
  o.x = fmaxf(fmaf(v.x, sc.x, sh.x), 0.0f);
  o.y = fmaxf(fmaf(v.y, sc.y, sh.y), 0.0f);
  o.z = fmaxf(fmaf(v.z, sc.z, sh.z), 0.0f);
  o.w = fmaxf(fmaf(v.w, sc.w, sh.w), 0.0f);
  out[i] = o;
}

// ---------------- graph sum-pool (4 elements / thread) ----------------
__global__ void k_graphpool(const float4* __restrict__ h, const int* __restrict__ gids,
                            float* __restrict__ gp, long long n4) {
  long long i = (long long)blockIdx.x * blockDim.x + threadIdx.x;
  if (i >= n4) return;
  long long e0 = i * 4;
  int node = (int)(e0 >> 7);
  int j = (int)(e0 & (HD - 1));
  float4 v = h[i];
  float* g = gp + (long long)gids[node] * HD + j;
  atomicAdd(g + 0, v.x);
  atomicAdd(g + 1, v.y);
  atomicAdd(g + 2, v.z);
  atomicAdd(g + 3, v.w);
}

__global__ void k_f2bf_arr(const float* __restrict__ in, unsigned short* __restrict__ out,
                           long long n) {
  long long i = (long long)blockIdx.x * blockDim.x + threadIdx.x;
  if (i < n) out[i] = f2bf(in[i]);
}

// ---------------- heads: score[g][o] = sum_l gp[l] @ Wp[l] + bp[l] ----------------
// gp bf16 [L][G][128], wpt bf16 [L][O][128], bp f32 [L][O]; one wave per 16x16 tile
__global__ void k_headgemm(const unsigned short* __restrict__ gp,
                           const unsigned short* __restrict__ wpt,
                           const float* __restrict__ bp,
                           float* __restrict__ out, int G, int O, int L) {
  int lane = threadIdx.x & 31;
  int r  = lane & 15;
  int hi = lane >> 4;
  int g0 = blockIdx.x * 16;
  int o0 = blockIdx.y * 16;
  v8f c = {};
  for (int l = 0; l < L; ++l) {
    const unsigned short* arow = gp  + ((long long)l * G + g0 + r) * HD;
    const unsigned short* brow = wpt + ((long long)l * O + o0 + r) * HD;
#pragma unroll
    for (int ks = 0; ks < 4; ++ks) {
      int k0 = ks * 32 + hi * 8;
      Frag16 a, b;
      a.q[0] = *(const uint4*)(arow + k0);
      a.q[1] = *(const uint4*)(arow + k0 + 16);
      b.q[0] = *(const uint4*)(brow + k0);
      b.q[1] = *(const uint4*)(brow + k0 + 16);
      c = __builtin_amdgcn_wmma_f32_16x16x32_bf16(false, a.v, false, b.v,
                                                  (short)0, c, false, false);
    }
  }
  int n = lane & 15;
  float bsum = 0.0f;
  for (int l = 0; l < L; ++l) bsum += bp[l * O + o0 + n];
#pragma unroll
  for (int i = 0; i < 8; ++i) {
    int m = i + hi * 8;
    out[(long long)(g0 + m) * O + o0 + n] = c[i] + bsum;
  }
}

// ---------------- host ----------------
extern "C" void kernel_launch(void* const* d_in, const int* in_sizes, int n_in,
                              void* d_out, int out_size, void* d_ws, size_t ws_size,
                              hipStream_t stream) {
  const float* x    = (const float*)d_in[0];
  const int*   src  = (const int*)  d_in[1];
  const int*   dst  = (const int*)  d_in[2];
  const int*   gids = (const int*)  d_in[3];
  const float* eps  = (const float*)d_in[5];
  const float* W1   = (const float*)d_in[6];
  const float* b1   = (const float*)d_in[7];
  const float* gm   = (const float*)d_in[8];
  const float* bm   = (const float*)d_in[9];
  const float* W2   = (const float*)d_in[10];
  const float* b2   = (const float*)d_in[11];
  const float* go   = (const float*)d_in[12];
  const float* bo   = (const float*)d_in[13];
  const float* Wp   = (const float*)d_in[14];
  const float* bp   = (const float*)d_in[15];

  const int N  = in_sizes[0] / HD;
  const int E  = in_sizes[1];
  const int LG = in_sizes[5];
  const int L  = LG + 1;
  const int O  = in_sizes[14] / (L * HD);
  const int G  = out_size / O;
  const long long NH  = (long long)N * HD;
  const long long NH4 = NH / 4;

  // workspace carve (256B aligned)
  char* ws = (char*)d_ws;
  size_t off = 0;
  auto carve = [&](size_t bytes) -> void* {
    off = (off + 255) & ~(size_t)255;
    void* p = ws + off;
    off += bytes;
    return p;
  };
  float*          agg   = (float*)         carve(NH * 4);
  float*          Z     = (float*)         carve(NH * 4);
  float*          hbuf  = (float*)         carve(NH * 4);
  unsigned short* stg   = (unsigned short*)carve(NH * 2);       // shared bf16 staging
  unsigned short* w1t   = (unsigned short*)carve((size_t)LG * HD * HD * 2);
  unsigned short* w2t   = (unsigned short*)carve((size_t)LG * HD * HD * 2);
  unsigned short* wpt   = (unsigned short*)carve((size_t)L * O * HD * 2);
  float*          gp    = (float*)         carve((size_t)L * G * HD * 4);
  unsigned short* gpbf  = (unsigned short*)carve((size_t)L * G * HD * 2);
  float*          sums  = (float*)         carve(256 * 4);      // sum[128] ++ sumsq[128]
  float*          scsh  = (float*)         carve(256 * 4);      // scale[128] ++ shift[128]
  (void)ws_size; (void)n_in;

  const int T = 256;
  const long long gpN  = (long long)L * G * HD;
  const long long gpN4 = gpN / 4;
  dim3 gemmGrid((N + 63) / 64, HD / 16);

  // convert + transpose weights to bf16
  {
    int t1 = LG * HD * HD;
    k_wt_bf16<<<(t1 + T - 1) / T, T, 0, stream>>>(W1, w1t, LG, HD, HD);
    k_wt_bf16<<<(t1 + T - 1) / T, T, 0, stream>>>(W2, w2t, LG, HD, HD);
    int t2 = L * HD * O;
    k_wt_bf16<<<(t2 + T - 1) / T, T, 0, stream>>>(Wp, wpt, L, HD, O);
  }

  // graph pool of layer-0 rep (the input features)
  k_zero4<<<(int)((gpN4 + T - 1) / T), T, 0, stream>>>((float4*)gp, gpN4);
  k_graphpool<<<(int)((NH4 + T - 1) / T), T, 0, stream>>>((const float4*)x, gids, gp, NH4);

  const float* hcur = x;
  const int rowsPerStatBlock = 512;
  const int statGrid = (N + rowsPerStatBlock - 1) / rowsPerStatBlock;
  const float invN = 1.0f / (float)N;

  for (int l = 0; l < LG; ++l) {
    // neighbor sum-pool
    k_zero4<<<(int)((NH4 + T - 1) / T), T, 0, stream>>>((float4*)agg, NH4);
    k_scatter<<<(int)(((long long)E * 32 + T - 1) / T), T, 0, stream>>>(hcur, src, dst, agg, E);
    // pooled = agg + (1+eps)*h -> bf16
    k_pooled_bf16<<<(int)((NH4 + T - 1) / T), T, 0, stream>>>(
        (const float4*)agg, (const float4*)hcur, eps, l, (ushort4*)stg, NH4);
    // Linear1 (WMMA)
    k_gemm128<<<gemmGrid, 32, 0, stream>>>(stg, w1t + (size_t)l * HD * HD, b1 + l * HD, Z, N);
    // inner BN -> ReLU -> bf16
    k_zero4<<<1, 64, 0, stream>>>((float4*)sums, 64);
    k_bnstats<<<statGrid, HD, 0, stream>>>(Z, sums, sums + HD, N, rowsPerStatBlock);
    k_bnfin<<<1, HD, 0, stream>>>(sums, sums + HD, gm + l * HD, bm + l * HD,
                                  scsh, scsh + HD, invN);
    k_bnrelu_bf16<<<(int)((NH4 + T - 1) / T), T, 0, stream>>>(
        (const float4*)Z, scsh, scsh + HD, (ushort4*)stg, NH4);
    // Linear2 (WMMA)
    k_gemm128<<<gemmGrid, 32, 0, stream>>>(stg, w2t + (size_t)l * HD * HD, b2 + l * HD, Z, N);
    // outer BN -> ReLU -> f32 h
    k_zero4<<<1, 64, 0, stream>>>((float4*)sums, 64);
    k_bnstats<<<statGrid, HD, 0, stream>>>(Z, sums, sums + HD, N, rowsPerStatBlock);
    k_bnfin<<<1, HD, 0, stream>>>(sums, sums + HD, go + l * HD, bo + l * HD,
                                  scsh, scsh + HD, invN);
    k_bnrelu_f32<<<(int)((NH4 + T - 1) / T), T, 0, stream>>>(
        (const float4*)Z, scsh, scsh + HD, (float4*)hbuf, NH4);
    // graph pool this layer's rep
    k_graphpool<<<(int)((NH4 + T - 1) / T), T, 0, stream>>>(
        (const float4*)hbuf, gids, gp + (size_t)(l + 1) * G * HD, NH4);
    hcur = hbuf;
  }

  // heads: cast pooled reps to bf16, fused WMMA over all L layers
  k_f2bf_arr<<<(int)((gpN + T - 1) / T), T, 0, stream>>>(gp, gpbf, gpN);
  dim3 headGrid(G / 16, O / 16);
  k_headgemm<<<headGrid, 32, 0, stream>>>(gpbf, wpt, bp, (float*)d_out, G, O, L);
}
